// GraphLevelGNNWithLSTM_51908974740046
// MI455X (gfx1250) — matrix operand order repaired and checked by reading, not compiled, not run on hardware
//
#include <hip/hip_runtime.h>
#include <stdint.h>

// ---------------- bf16 helpers ----------------
__device__ __forceinline__ uint16_t f2bf(float f) {
    uint32_t u = __float_as_uint(f);
    uint32_t r = u + 0x7FFFu + ((u >> 16) & 1u);   // round-to-nearest-even
    return (uint16_t)(r >> 16);
}

typedef __attribute__((ext_vector_type(16))) __bf16 v16bf;
typedef __attribute__((ext_vector_type(8)))  float  v8f;

union AFrag { v16bf v; uint32_t u[8]; };

// ---------------- elementwise utility kernels ----------------
__global__ void fill_f32_kernel(float* p, float v, int n) {
    int i = blockIdx.x * blockDim.x + threadIdx.x;
    if (i < n) p[i] = v;
}

__global__ void deg_accum_kernel(const int* dst, float* deg, int E) {
    int e = blockIdx.x * blockDim.x + threadIdx.x;
    if (e < E) atomicAdd(&deg[dst[e]], 1.0f);
}

__global__ void rsqrt_kernel(const float* deg, float* dinv, int n) {
    int i = blockIdx.x * blockDim.x + threadIdx.x;
    if (i < n) dinv[i] = rsqrtf(deg[i]);
}

__global__ void cast_bf16_kernel(const float* x, uint16_t* y, int n) {
    int i = blockIdx.x * blockDim.x + threadIdx.x;
    if (i < n) y[i] = f2bf(x[i]);
}

// Pack a weight matrix into column-major-of-B bf16:  Bt[n*Kp + k] = W[k*sK + n*sN]
// (zero padded for k >= Kv or n >= Nv)
__global__ void pack_bt_kernel(const float* W, uint16_t* Bt,
                               int sK, int sN, int Kv, int Nv, int Kp, int Np) {
    int i = blockIdx.x * blockDim.x + threadIdx.x;
    if (i >= Kp * Np) return;
    int n = i / Kp, k = i % Kp;
    float v = (k < Kv && n < Nv) ? W[(size_t)k * sK + (size_t)n * sN] : 0.0f;
    Bt[(size_t)n * Kp + k] = f2bf(v);
}

// ---------------- WMMA bf16 GEMM: C[M x N] = A[M x K] * B, B stored col-major Bt[N][K]
// block = 128 (4 waves), each wave computes one 16x16 C tile; 4 M-tiles per block.
__global__ void gemm_bf16_wmma_kernel(const uint16_t* __restrict__ A,
                                      const uint16_t* __restrict__ Bt,
                                      float* __restrict__ C,
                                      int M, int lda, int ldb, int ldc, int ksteps) {
    const int lane = threadIdx.x & 31;
    const int wave = threadIdx.x >> 5;
    const int tm   = blockIdx.x * 4 + wave;        // M tile index
    const int tn   = blockIdx.y;                   // N tile index
    if (tm * 16 >= M) return;

    const int m   = lane & 15;                     // row within tile (A) / col (B,C)
    const int sel = lane >> 4;                     // lane-group select
    int arow = tm * 16 + m;
    if (arow >= M) arow = M - 1;                   // clamp (store is guarded)

    const uint16_t* __restrict__ ap = A  + (size_t)arow * lda;
    const uint16_t* __restrict__ bp = Bt + (size_t)(tn * 16 + m) * ldb;
    const int kbA = sel * 16;                      // A: lanes 16-31 hold K+16 block
    const int kbB = sel * 16;                      // B: lanes 16-31 hold K+16 block

    v8f acc = {};
    for (int ks = 0; ks < ksteps; ++ks) {
        const int k0 = ks * 32;
        AFrag a, b;
        #pragma unroll
        for (int j = 0; j < 4; ++j)
            a.u[j] = *reinterpret_cast<const uint32_t*>(ap + k0 + kbA + 2 * j);
        #pragma unroll
        for (int j = 0; j < 4; ++j)
            a.u[4 + j] = *reinterpret_cast<const uint32_t*>(ap + k0 + 8 + kbA + 2 * j);
        #pragma unroll
        for (int j = 0; j < 8; ++j)
            b.u[j] = *reinterpret_cast<const uint32_t*>(bp + k0 + kbB + 2 * j);
        acc = __builtin_amdgcn_wmma_f32_16x16x32_bf16(
            false, a.v, false, b.v, (short)0, acc, false, false);
    }

    // C layout: VGPR r -> row (sel*8 + r), col = lane&15
    const int row0 = tm * 16 + sel * 8;
    const int col  = tn * 16 + m;
    #pragma unroll
    for (int r = 0; r < 8; ++r) {
        int row = row0 + r;
        if (row < M) C[(size_t)row * ldc + col] = acc[r];
    }
}

// ---------------- edge scatter:  agg[dst] += dinv[src]*dinv[dst] * h[src]  ----------------
__global__ void scatter_kernel(const int* __restrict__ src, const int* __restrict__ dst,
                               const float* __restrict__ dinv, const float* __restrict__ h,
                               float* __restrict__ agg, int E, int F4, int ldf) {
    int tid = blockIdx.x * blockDim.x + threadIdx.x;
    if (tid >= E * F4) return;
    int e = tid / F4, c = tid % F4;
    int s = src[e], d = dst[e];
    float norm = dinv[s] * dinv[d];
    const float4 v = *reinterpret_cast<const float4*>(&h[(size_t)s * ldf + c * 4]);
    float* o = &agg[(size_t)d * ldf + c * 4];
    atomicAdd(o + 0, norm * v.x);
    atomicAdd(o + 1, norm * v.y);
    atomicAdd(o + 2, norm * v.z);
    atomicAdd(o + 3, norm * v.w);
}

// ---------------- self-loop + bias + ReLU + bf16 recast (padded) ----------------
__global__ void finish_kernel(const float* __restrict__ agg, const float* __restrict__ h,
                              const float* __restrict__ dinv, const float* __restrict__ bias,
                              uint16_t* __restrict__ out_bf, int N, int Fvalid,
                              int ldin, int ldout) {
    int tid = blockIdx.x * blockDim.x + threadIdx.x;
    if (tid >= N * ldout) return;
    int i = tid / ldout, f = tid % ldout;
    float v = 0.0f;
    if (f < Fvalid) {
        float di = dinv[i];
        v = agg[(size_t)i * ldin + f] + di * di * h[(size_t)i * ldin + f] + bias[f];
        v = fmaxf(v, 0.0f);
    }
    out_bf[(size_t)i * ldout + f] = f2bf(v);
}

// ---------------- sequential LSTM scan (batch=1, H=32, gates precomputed) ------------
__global__ void lstm_scan_kernel(const float* __restrict__ gates_x,   // [T][128]
                                 const float* __restrict__ w_hh,      // [128][32]
                                 const float* __restrict__ b_ih, const float* __restrict__ b_hh,
                                 const float* __restrict__ fc_w, const float* __restrict__ fc_b,
                                 int T, float* __restrict__ out) {
    const int j = threadIdx.x;                  // 0..127, one gate output per thread
    __shared__ float h_s[32], c_s[32], g_s[128];

    float wr[32];
    #pragma unroll
    for (int k = 0; k < 32; ++k) wr[k] = w_hh[j * 32 + k];
    const float bj = b_ih[j] + b_hh[j];

    if (j < 32) { h_s[j] = 0.0f; c_s[j] = 0.0f; }
    __syncthreads();

    for (int t = 0; t < T; ++t) {
        float g = gates_x[(size_t)t * 128 + j] + bj;
        #pragma unroll
        for (int k = 0; k < 32; ++k) g += wr[k] * h_s[k];
        g_s[j] = g;
        __syncthreads();
        if (j < 32) {
            float ig = 1.0f / (1.0f + __expf(-g_s[j]));
            float fg = 1.0f / (1.0f + __expf(-g_s[32 + j]));
            float gg = tanhf(g_s[64 + j]);
            float og = 1.0f / (1.0f + __expf(-g_s[96 + j]));
            float c  = fg * c_s[j] + ig * gg;
            c_s[j] = c;
            h_s[j] = og * tanhf(c);
        }
        __syncthreads();
    }

    if (j == 0) {
        float acc = fc_b[0];
        for (int k = 0; k < 32; ++k) acc += fc_w[k] * h_s[k];
        out[0] = acc;
    }
}

// ---------------- host side ----------------
static inline int cdiv(int a, int b) { return (a + b - 1) / b; }

extern "C" void kernel_launch(void* const* d_in, const int* in_sizes, int n_in,
                              void* d_out, int out_size, void* d_ws, size_t ws_size,
                              hipStream_t stream) {
    const float* x    = (const float*)d_in[0];
    const int*   ei   = (const int*)  d_in[1];
    const float* W1   = (const float*)d_in[2];
    const float* b1   = (const float*)d_in[3];
    const float* W2   = (const float*)d_in[4];
    const float* b2   = (const float*)d_in[5];
    const float* w_ih = (const float*)d_in[6];
    const float* w_hh = (const float*)d_in[7];
    const float* b_ih = (const float*)d_in[8];
    const float* b_hh = (const float*)d_in[9];
    const float* fc_w = (const float*)d_in[10];
    const float* fc_b = (const float*)d_in[11];
    float* out = (float*)d_out;

    const int IN_CH = 256, H1 = 71, H2 = 82;
    const int H1P = 80;   // layer-1 width padded to 16
    const int KP2 = 96;   // K padding for GEMM2/GEMM3 (multiple of 32)
    const int H2P = 96;   // layer-2 width padded
    const int G   = 128;  // 4*LSTM_H gate width

    const int N = in_sizes[0] / IN_CH;     // 20000
    const int E = in_sizes[1] / 2;         // 640000
    const int* srcp = ei;
    const int* dstp = ei + E;

    // workspace carve-out (256B aligned)
    char* base = (char*)d_ws;
    size_t off = 0;
    auto carve = [&](size_t bytes) -> char* {
        char* p = base + off;
        off = (off + bytes + 255) & ~(size_t)255;
        return p;
    };
    float*    deg    = (float*)   carve((size_t)N * 4);
    float*    dinv   = (float*)   carve((size_t)N * 4);
    uint16_t* x_bf   = (uint16_t*)carve((size_t)N * IN_CH * 2);
    uint16_t* w1t    = (uint16_t*)carve((size_t)H1P * IN_CH * 2);
    float*    h1     = (float*)   carve((size_t)N * H1P * 4);
    float*    agg1   = (float*)   carve((size_t)N * H1P * 4);
    uint16_t* h1r_bf = (uint16_t*)carve((size_t)N * KP2 * 2);
    uint16_t* w2t    = (uint16_t*)carve((size_t)H2P * KP2 * 2);
    float*    h2     = (float*)   carve((size_t)N * H2P * 4);
    float*    agg2   = (float*)   carve((size_t)N * H2P * 4);
    uint16_t* h2r_bf = (uint16_t*)carve((size_t)N * KP2 * 2);
    uint16_t* wiht   = (uint16_t*)carve((size_t)G * KP2 * 2);
    float*    gates  = (float*)   carve((size_t)N * G * 4);
    (void)ws_size;

    const int BT = 256;
    const int mtiles = cdiv(N, 16);

    // --- degree / normalization ---
    fill_f32_kernel<<<cdiv(N, BT), BT, 0, stream>>>(deg, 1.0f, N);   // self-loop
    deg_accum_kernel<<<cdiv(E, BT), BT, 0, stream>>>(dstp, deg, E);
    rsqrt_kernel<<<cdiv(N, BT), BT, 0, stream>>>(deg, dinv, N);

    // --- operand prep (bf16 cast / pad / B pre-transpose) ---
    cast_bf16_kernel<<<cdiv(N * IN_CH, BT), BT, 0, stream>>>(x, x_bf, N * IN_CH);
    pack_bt_kernel<<<cdiv(H1P * IN_CH, BT), BT, 0, stream>>>(W1, w1t, H1, 1, IN_CH, H1, IN_CH, H1P);
    pack_bt_kernel<<<cdiv(H2P * KP2, BT), BT, 0, stream>>>(W2, w2t, H2, 1, H1, H2, KP2, H2P);
    pack_bt_kernel<<<cdiv(G * KP2, BT), BT, 0, stream>>>(w_ih, wiht, 1, H2, H2, G, KP2, G);

    // --- GCN layer 1 ---
    gemm_bf16_wmma_kernel<<<dim3(cdiv(mtiles, 4), H1P / 16), 128, 0, stream>>>(
        x_bf, w1t, h1, N, IN_CH, IN_CH, H1P, IN_CH / 32);
    fill_f32_kernel<<<cdiv(N * H1P, BT), BT, 0, stream>>>(agg1, 0.0f, N * H1P);
    scatter_kernel<<<cdiv(E * (H1P / 4), BT), BT, 0, stream>>>(
        srcp, dstp, dinv, h1, agg1, E, H1P / 4, H1P);
    finish_kernel<<<cdiv(N * KP2, BT), BT, 0, stream>>>(
        agg1, h1, dinv, b1, h1r_bf, N, H1, H1P, KP2);

    // --- GCN layer 2 ---
    gemm_bf16_wmma_kernel<<<dim3(cdiv(mtiles, 4), H2P / 16), 128, 0, stream>>>(
        h1r_bf, w2t, h2, N, KP2, KP2, H2P, KP2 / 32);
    fill_f32_kernel<<<cdiv(N * H2P, BT), BT, 0, stream>>>(agg2, 0.0f, N * H2P);
    scatter_kernel<<<cdiv(E * (H2P / 4), BT), BT, 0, stream>>>(
        srcp, dstp, dinv, h2, agg2, E, H2P / 4, H2P);
    finish_kernel<<<cdiv(N * KP2, BT), BT, 0, stream>>>(
        agg2, h2, dinv, b2, h2r_bf, N, H2, H2P, KP2);

    // --- LSTM input projection: gates_x = h2r @ w_ih^T  (WMMA) ---
    gemm_bf16_wmma_kernel<<<dim3(cdiv(mtiles, 4), G / 16), 128, 0, stream>>>(
        h2r_bf, wiht, gates, N, KP2, KP2, G, KP2 / 32);

    // --- sequential LSTM scan + final FC ---
    lstm_scan_kernel<<<1, 128, 0, stream>>>(gates, w_hh, b_ih, b_hh, fc_w, fc_b, N, out);
}
